// MEGConv_75788992906320
// MI455X (gfx1250) — matrix operand order repaired and checked by reading, not compiled, not run on hardware
//
#include <hip/hip_runtime.h>
#include <hip/hip_bf16.h>

typedef __attribute__((ext_vector_type(16))) _Float16 v16h;
typedef __attribute__((ext_vector_type(8)))  _Float16 v8h;
typedef __attribute__((ext_vector_type(8)))  float    v8f;

#define DFEAT 128

// ---------------------------------------------------------------- helpers

__device__ __forceinline__ float softplus_f(float x) {
    // stable: max(x,0) + log1p(exp(-|x|))
    float ax = fabsf(x);
    return fmaxf(x, 0.0f) + log1pf(__expf(-ax));
}

// A-fragment (16x32 f16) from a row-major [16][K] tile.
// Lane layout (ISA 7.12.2): lane m in 0..15 (row M=m), group g = lane>>4.
// halves 0..7  -> K = kb*32 + 8g + i        (contiguous 16B)
// halves 8..15 -> K = kb*32 + 16 + 8g + i   (contiguous 16B)
__device__ __forceinline__ v16h load_a_frag(const _Float16* rowk, int g) {
    v8h lo = *(const v8h*)(rowk + 8 * g);
    v8h hi = *(const v8h*)(rowk + 16 + 8 * g);
    v16h a;
#pragma unroll
    for (int i = 0; i < 8; ++i) { a[i] = lo[i]; a[i + 8] = hi[i]; }
    return a;
}

// One 16x16 output tile of  D = X[16xK] * W[Kx128] + bias, accumulated f32.
// wT is the transposed f16 weight: wT[n][k] = w[k][n], row-major [128][K],
// so the B-fragment (32x16): lane n = column, same K order as the A-fragment
// (two contiguous 16-byte loads per lane).
template<int K>
__device__ __forceinline__ v8f gemm_tile(const _Float16* xs, const _Float16* wT,
                                         const float* bias, int ncol, int m, int g) {
    v8f acc;
    float b0 = bias[ncol];
#pragma unroll
    for (int r = 0; r < 8; ++r) acc[r] = b0;
    const _Float16* wrow = wT + (size_t)ncol * K;
    const _Float16* xrow = xs + m * K;
#pragma unroll
    for (int kb = 0; kb < K / 32; ++kb) {
        v16h a = load_a_frag(xrow + kb * 32, g);
        v16h b = load_a_frag(wrow + kb * 32, g);
        acc = __builtin_amdgcn_wmma_f32_16x16x32_f16(false, a, false, b,
                                                     (short)0, acc, false, false);
    }
    return acc;
}

// 3-layer MLP on a 16-row tile staged in LDS (K0 -> 128 -> 128 -> 128),
// softplus on layers 0,1; identity on layer 2.  blockDim.x must be 256.
template<int K0>
__device__ __forceinline__ void mlp3(const _Float16* xs, _Float16* h1, _Float16* h2,
                                     const _Float16* w0T, const float* b0,
                                     const _Float16* w1T, const float* b1,
                                     const _Float16* w2T, const float* b2,
                                     float* outrow, int rowlim) {
    const int tid  = threadIdx.x;
    const int wave = tid >> 5, lane = tid & 31;
    const int g = lane >> 4, m = lane & 15;
    const int ncol = (wave << 4) + m;          // 0..127 output column

    v8f acc = gemm_tile<K0>(xs, w0T, b0, ncol, m, g);
#pragma unroll
    for (int r = 0; r < 8; ++r)                // C/D layout: M = 8g + r, N = ncol
        h1[(8 * g + r) * DFEAT + ncol] = (_Float16)softplus_f(acc[r]);
    __syncthreads();

    acc = gemm_tile<DFEAT>(h1, w1T, b1, ncol, m, g);
#pragma unroll
    for (int r = 0; r < 8; ++r)
        h2[(8 * g + r) * DFEAT + ncol] = (_Float16)softplus_f(acc[r]);
    __syncthreads();

    acc = gemm_tile<DFEAT>(h2, w2T, b2, ncol, m, g);
#pragma unroll
    for (int r = 0; r < 8; ++r) {
        int row = 8 * g + r;
        if (row < rowlim)
            outrow[(size_t)row * DFEAT + ncol] = acc[r];
    }
}

// ---------------------------------------------------------------- kernels

// f32 (fi,fo) row-major -> f16 transposed [fo][fi]
__global__ __launch_bounds__(256)
void convert_wT(const float* __restrict__ w, _Float16* __restrict__ wT, int fi, int fo) {
    int idx = blockIdx.x * 256 + threadIdx.x;
    if (idx >= fi * fo) return;
    int k = idx % fi, n = idx / fi;
    wT[idx] = (_Float16)w[(size_t)k * fo + n];
}

// bond features scattered to both endpoint atoms + degree counts
__global__ __launch_bounds__(256)
void scatter_bond_kernel(const float* __restrict__ bond_feats, const int* __restrict__ bond_atom,
                         float* __restrict__ msum, float* __restrict__ deg, int nb) {
    long long idx = (long long)blockIdx.x * 256 + threadIdx.x;   // over 2*nb*32
    if (idx >= (long long)nb * 64) return;
    int q = (int)(idx & 31);                 // 4-float chunk within the 128-dim row
    long long e = idx >> 5;                  // flat edge slot 0..2nb-1
    int b = (int)(e >> 1);
    int atom = bond_atom[e];
    float4 v = *(const float4*)(bond_feats + (size_t)b * DFEAT + q * 4);
    float* dst = msum + (size_t)atom * DFEAT + q * 4;
    atomicAdd(dst + 0, v.x); atomicAdd(dst + 1, v.y);
    atomicAdd(dst + 2, v.z); atomicAdd(dst + 3, v.w);
    if (q == 0) atomicAdd(deg + atom, 1.0f);
}

// generic per-segment sum + count (for the global-node means)
__global__ __launch_bounds__(256)
void scatter_seg_kernel(const float* __restrict__ feats, const int* __restrict__ seg,
                        float* __restrict__ sum, float* __restrict__ cnt, int n) {
    long long idx = (long long)blockIdx.x * 256 + threadIdx.x;   // over n*32
    if (idx >= (long long)n * 32) return;
    int q = (int)(idx & 31);
    int i = (int)(idx >> 5);
    int s = seg[i];
    float4 v = *(const float4*)(feats + (size_t)i * DFEAT + q * 4);
    float* dst = sum + (size_t)s * DFEAT + q * 4;
    atomicAdd(dst + 0, v.x); atomicAdd(dst + 1, v.y);
    atomicAdd(dst + 2, v.z); atomicAdd(dst + 3, v.w);
    if (q == 0) atomicAdd(cnt + s, 1.0f);
}

__global__ __launch_bounds__(256)
void atom_mlp_kernel(const float* __restrict__ atom_feats, const float* __restrict__ msum,
                     const float* __restrict__ deg, const float* __restrict__ gfeat,
                     const int* __restrict__ atom_mol,
                     const _Float16* __restrict__ w0T, const float* __restrict__ b0,
                     const _Float16* __restrict__ w1T, const float* __restrict__ b1,
                     const _Float16* __restrict__ w2T, const float* __restrict__ b2,
                     float* __restrict__ out, int n) {
    __shared__ _Float16 xs[16 * 384];
    __shared__ _Float16 h1[16 * DFEAT];
    __shared__ _Float16 h2[16 * DFEAT];
    int row0 = blockIdx.x * 16;
    for (int idx = threadIdx.x; idx < 16 * 384; idx += 256) {
        int r = idx / 384, c = idx % 384;
        int a = row0 + r; if (a >= n) a = n - 1;
        float v;
        if (c < 128)      v = atom_feats[(size_t)a * DFEAT + c];
        else if (c < 256) {
            float s = msum[(size_t)a * DFEAT + (c - 128)];
            float d = deg[a];
            v = (d > 0.0f) ? (s / d) : 0.0f;
        } else            v = gfeat[(size_t)atom_mol[a] * DFEAT + (c - 256)];
        xs[idx] = (_Float16)v;
    }
    __syncthreads();
    mlp3<384>(xs, h1, h2, w0T, b0, w1T, b1, w2T, b2, out + (size_t)row0 * DFEAT, n - row0);
}

__global__ __launch_bounds__(256)
void bond_mlp_kernel(const float* __restrict__ bond_feats, const float* __restrict__ atom_out,
                     const float* __restrict__ gfeat,
                     const int* __restrict__ bond_atom, const int* __restrict__ bond_mol,
                     const _Float16* __restrict__ w0T, const float* __restrict__ b0,
                     const _Float16* __restrict__ w1T, const float* __restrict__ b1,
                     const _Float16* __restrict__ w2T, const float* __restrict__ b2,
                     float* __restrict__ out, int n) {
    __shared__ _Float16 xs[16 * 512];
    __shared__ _Float16 h1[16 * DFEAT];
    __shared__ _Float16 h2[16 * DFEAT];
    int row0 = blockIdx.x * 16;
    for (int idx = threadIdx.x; idx < 16 * 512; idx += 256) {
        int r = idx / 512, c = idx % 512;
        int b = row0 + r; if (b >= n) b = n - 1;
        float v;
        if (c < 128)      v = bond_feats[(size_t)b * DFEAT + c];
        else if (c < 256) v = atom_out[(size_t)bond_atom[b * 2 + 0] * DFEAT + (c - 128)];
        else if (c < 384) v = atom_out[(size_t)bond_atom[b * 2 + 1] * DFEAT + (c - 256)];
        else              v = gfeat[(size_t)bond_mol[b] * DFEAT + (c - 384)];
        xs[idx] = (_Float16)v;
    }
    __syncthreads();
    mlp3<512>(xs, h1, h2, w0T, b0, w1T, b1, w2T, b2, out + (size_t)row0 * DFEAT, n - row0);
}

__global__ __launch_bounds__(256)
void glob_mlp_kernel(const float* __restrict__ gfeat,
                     const float* __restrict__ gsum_a, const float* __restrict__ gcnt_a,
                     const float* __restrict__ gsum_b, const float* __restrict__ gcnt_b,
                     const _Float16* __restrict__ w0T, const float* __restrict__ b0,
                     const _Float16* __restrict__ w1T, const float* __restrict__ b1,
                     const _Float16* __restrict__ w2T, const float* __restrict__ b2,
                     float* __restrict__ out, int n) {
    __shared__ _Float16 xs[16 * 384];
    __shared__ _Float16 h1[16 * DFEAT];
    __shared__ _Float16 h2[16 * DFEAT];
    int row0 = blockIdx.x * 16;
    for (int idx = threadIdx.x; idx < 16 * 384; idx += 256) {
        int r = idx / 384, c = idx % 384;
        int g = row0 + r; if (g >= n) g = n - 1;
        float v;
        if (c < 128)      v = gfeat[(size_t)g * DFEAT + c];
        else if (c < 256) {
            float ca = gcnt_a[g];
            v = (ca > 0.0f) ? (gsum_a[(size_t)g * DFEAT + (c - 128)] / ca) : 0.0f;
        } else {
            float cb = gcnt_b[g];
            v = (cb > 0.0f) ? (gsum_b[(size_t)g * DFEAT + (c - 256)] / cb) : 0.0f;
        }
        xs[idx] = (_Float16)v;
    }
    __syncthreads();
    mlp3<384>(xs, h1, h2, w0T, b0, w1T, b1, w2T, b2, out + (size_t)row0 * DFEAT, n - row0);
}

// ---------------------------------------------------------------- launch

extern "C" void kernel_launch(void* const* d_in, const int* in_sizes, int n_in,
                              void* d_out, int out_size, void* d_ws, size_t ws_size,
                              hipStream_t stream) {
    const float* atom_feats = (const float*)d_in[0];
    const float* bond_feats = (const float*)d_in[1];
    const float* glob_feats = (const float*)d_in[2];
    const int*   bond_atom  = (const int*)d_in[3];
    const int*   atom_mol   = (const int*)d_in[4];
    const int*   bond_mol   = (const int*)d_in[5];
    const float* W[9]; const float* B[9];
    for (int i = 0; i < 9; ++i) { W[i] = (const float*)d_in[6 + 2 * i]; B[i] = (const float*)d_in[7 + 2 * i]; }

    const int na = in_sizes[0] / DFEAT;
    const int nb = in_sizes[1] / DFEAT;
    const int ng = in_sizes[2] / DFEAT;

    float* out      = (float*)d_out;
    float* atom_out = out;
    float* bond_out = out + (size_t)na * DFEAT;
    float* glob_out = bond_out + (size_t)nb * DFEAT;
    float* msum     = bond_out;   // reuse bond-out region as bond->atom sum scratch

    // workspace: 9 transposed f16 weights, then f32 scatter buffers
    static const int FI[9] = {3 * DFEAT, DFEAT, DFEAT, 4 * DFEAT, DFEAT, DFEAT, 3 * DFEAT, DFEAT, DFEAT};
    _Float16* hws = (_Float16*)d_ws;
    _Float16* wT[9]; size_t off = 0;
    for (int i = 0; i < 9; ++i) { wT[i] = hws + off; off += (size_t)FI[i] * DFEAT; }
    float* fws    = (float*)(hws + off);
    float* deg    = fws;                          // na
    float* gsum_a = deg + na;                     // ng*128
    float* gcnt_a = gsum_a + (size_t)ng * DFEAT;  // ng
    float* gsum_b = gcnt_a + ng;                  // ng*128
    float* gcnt_b = gsum_b + (size_t)ng * DFEAT;  // ng
    float* fend   = gcnt_b + ng;

    for (int i = 0; i < 9; ++i) {
        int nEl = FI[i] * DFEAT;
        convert_wT<<<(nEl + 255) / 256, 256, 0, stream>>>(W[i], wT[i], FI[i], DFEAT);
    }
    hipMemsetAsync(deg, 0, (char*)fend - (char*)deg, stream);
    hipMemsetAsync(msum, 0, (size_t)na * DFEAT * sizeof(float), stream);

    {
        long long tot = (long long)nb * 64;
        scatter_bond_kernel<<<(unsigned)((tot + 255) / 256), 256, 0, stream>>>(
            bond_feats, bond_atom, msum, deg, nb);
    }

    atom_mlp_kernel<<<(na + 15) / 16, 256, 0, stream>>>(
        atom_feats, msum, deg, glob_feats, atom_mol,
        wT[0], B[0], wT[1], B[1], wT[2], B[2], atom_out, na);

    {
        long long tot = (long long)na * 32;
        scatter_seg_kernel<<<(unsigned)((tot + 255) / 256), 256, 0, stream>>>(
            atom_out, atom_mol, gsum_a, gcnt_a, na);
    }

    bond_mlp_kernel<<<(nb + 15) / 16, 256, 0, stream>>>(
        bond_feats, atom_out, glob_feats, bond_atom, bond_mol,
        wT[3], B[3], wT[4], B[4], wT[5], B[5], bond_out, nb);

    {
        long long tot = (long long)nb * 32;
        scatter_seg_kernel<<<(unsigned)((tot + 255) / 256), 256, 0, stream>>>(
            bond_out, bond_mol, gsum_b, gcnt_b, nb);
    }

    glob_mlp_kernel<<<(ng + 15) / 16, 256, 0, stream>>>(
        glob_feats, gsum_a, gcnt_a, gsum_b, gcnt_b,
        wT[6], B[6], wT[7], B[7], wT[8], B[8], glob_out, ng);
}
